// route_func_linear_cls_token_moe_42039139893873
// MI455X (gfx1250) — matrix operand order repaired and checked by reading, not compiled, not run on hardware
//
#include <hip/hip_runtime.h>
#include <hip/hip_bf16.h>

typedef _Float16 f16;
typedef __attribute__((ext_vector_type(16))) _Float16 v16h;
typedef __attribute__((ext_vector_type(8)))  _Float16 v8h;
typedef __attribute__((ext_vector_type(8)))  float    v8f;
typedef __attribute__((ext_vector_type(4)))  int      v4i;

// Problem constants
#define NB    64
#define CIN   512
#define TT    16
#define COUT  512
#define KW    3
#define NE    16
#define KTOT  (NE*CIN*KW)   // 24576
#define NTOT  (NB*TT)       // 1024

// CDNA5 async global->LDS path (guarded so both toolchains compile)
#if defined(__has_builtin)
#if __has_builtin(__builtin_amdgcn_global_load_async_to_lds_b128) && \
    __has_builtin(__builtin_amdgcn_s_wait_asynccnt)
#define ASYNC_OK 1
#endif
#endif
#ifndef ASYNC_OK
#define ASYNC_OK 0
#endif

__device__ __forceinline__ void copy16B_to_lds(const f16* g, f16* l)
{
#if ASYNC_OK
    __builtin_amdgcn_global_load_async_to_lds_b128(
        (__attribute__((address_space(1))) v4i*)g,
        (__attribute__((address_space(3))) v4i*)l, 0, 0);
#else
    *(v8h*)l = *(const v8h*)g;
#endif
}

// ---------------------------------------------------------------------------
// Kernel 1: pooled -> rf (B,E) and bb = rf @ bias (B,COUT)
// ---------------------------------------------------------------------------
__global__ __launch_bounds__(256) void k_rf_bb(const float* __restrict__ x,
                                               const float* __restrict__ rf_w,
                                               const float* __restrict__ rf_b,
                                               const float* __restrict__ bias,
                                               float* __restrict__ rf_out,
                                               float* __restrict__ bb_out)
{
    __shared__ float pooled[CIN];
    __shared__ float rf_s[NE];
    const int b   = blockIdx.x;
    const int tid = threadIdx.x;

    for (int c = tid; c < CIN; c += 256) {
        const float* xp = x + ((long)b*CIN + c)*TT;
        float s = 0.f;
        #pragma unroll
        for (int t = 0; t < TT; ++t) s += xp[t];
        pooled[c] = s * (1.0f/16.0f);
    }
    __syncthreads();

    if (tid < NE) {
        float s = rf_b[tid];
        const float* w = rf_w + tid*CIN;
        for (int c = 0; c < CIN; ++c) s += pooled[c]*w[c];
        rf_s[tid] = s;
        rf_out[b*NE + tid] = s;
    }
    __syncthreads();

    for (int o = tid; o < COUT; o += 256) {
        float s = 0.f;
        #pragma unroll
        for (int e = 0; e < NE; ++e) s += rf_s[e] * bias[e*COUT + o];
        bb_out[b*COUT + o] = s;
    }
}

// ---------------------------------------------------------------------------
// Kernel 2: weight f32 [e][o][i*3+k]  ->  Ah f16 [o][e*1536 + i*3 + k]
// ---------------------------------------------------------------------------
__global__ __launch_bounds__(256) void k_prep_a(const float* __restrict__ w,
                                                f16* __restrict__ Ah)
{
    const long tid = (long)blockIdx.x*256 + threadIdx.x;
    const long d   = tid*8;
    const int  o   = (int)(d / KTOT);
    const int  f   = (int)(d % KTOT);
    const int  e   = f / (CIN*KW);
    const int  g   = f % (CIN*KW);
    const float* src = w + ((long)(e*COUT + o))*(CIN*KW) + g;
    v8h outv;
    #pragma unroll
    for (int j = 0; j < 8; ++j) outv[j] = (f16)src[j];
    *(v8h*)(Ah + d) = outv;
}

// ---------------------------------------------------------------------------
// Kernel 3: Bt f16 [n=b*16+t][e*1536+i*3+k] = rf[b,e] * x_pad[b,i,t+k-1]
// ---------------------------------------------------------------------------
__global__ __launch_bounds__(256) void k_prep_b(const float* __restrict__ x,
                                                const float* __restrict__ rf,
                                                f16* __restrict__ Bt)
{
    const long tid = (long)blockIdx.x*256 + threadIdx.x;
    const long d   = tid*8;
    const int  n   = (int)(d / KTOT);
    const int  f   = (int)(d % KTOT);
    const int  b   = n >> 4;
    const int  t   = n & 15;
    const int  e   = f / (CIN*KW);
    const int  g   = f % (CIN*KW);
    const float rfv = rf[b*NE + e];
    v8h outv;
    #pragma unroll
    for (int j = 0; j < 8; ++j) {
        const int gg = g + j;
        const int i  = gg / 3;
        const int kz = gg - i*3;
        const int ts = t + kz - 1;
        float val = (ts >= 0 && ts < TT) ? x[(((long)b*CIN + i) << 4) + ts] : 0.0f;
        outv[j] = (f16)(rfv * val);
    }
    *(v8h*)(Bt + d) = outv;
}

// ---------------------------------------------------------------------------
// Kernel 4: GEMM  out[o, n] = Ah[o,:] . Bt[n,:]  + bb + 1
//   M=512, N=1024, K=24576 ; 64x64 tiles, 8 waves, async-LDS double buffer
// ---------------------------------------------------------------------------
#define KSTEP  64
#define LDA    72                 // padded LDS row stride in halves (16 bank groups)
#define NSTAGE (KTOT/KSTEP)       // 384

__global__ __launch_bounds__(256) void k_gemm(const f16* __restrict__ Ah,
                                              const f16* __restrict__ Bt,
                                              const float* __restrict__ bb,
                                              float* __restrict__ out)
{
    __shared__ __align__(16) f16 sA[2][64*LDA];
    __shared__ __align__(16) f16 sB[2][64*LDA];

    const int mBlk = blockIdx.x;            // 0..7
    const int nBlk = blockIdx.y;            // 0..15
    const int tid  = threadIdx.x;
    const int wave = tid >> 5;              // wave32
    const int lane = tid & 31;
    const int lh   = lane >> 4;
    const int lr   = lane & 15;
    const int ni   = wave & 3;
    const int mi0  = (wave >> 2) * 2;

    // Producer mapping: 512 chunks of 16B per operand tile, 2 per thread.
    const int ar0 = tid >> 3,          ac0 = (tid & 7) * 8;           // halves
    const int ar1 = (tid + 256) >> 3,  ac1 = ((tid + 256) & 7) * 8;

    const f16* gA = Ah + (long)(mBlk*64)*KTOT;
    const f16* gB = Bt + (long)(nBlk*64)*KTOT;

    auto issue = [&](int kk, int buf) {
        copy16B_to_lds(gA + (long)ar0*KTOT + kk + ac0, &sA[buf][ar0*LDA + ac0]);
        copy16B_to_lds(gA + (long)ar1*KTOT + kk + ac1, &sA[buf][ar1*LDA + ac1]);
        copy16B_to_lds(gB + (long)ar0*KTOT + kk + ac0, &sB[buf][ar0*LDA + ac0]);
        copy16B_to_lds(gB + (long)ar1*KTOT + kk + ac1, &sB[buf][ar1*LDA + ac1]);
    };

    v8f acc0 = {};
    v8f acc1 = {};

    auto consume = [&](int buf) {
        const f16* A0 = &sA[buf][(mi0*16       + lr)*LDA + lh*8];
        const f16* A1 = &sA[buf][((mi0 + 1)*16 + lr)*LDA + lh*8];
        const f16* Bp = &sB[buf][(ni*16        + lr)*LDA + lh*8];
        #pragma unroll
        for (int k2 = 0; k2 < KSTEP; k2 += 32) {
            v16h a0, a1, bf;
            *((v8h*)&a0)     = *(const v8h*)(A0 + k2);        // K 0..7   (lh=0) / 8..15  (lh=1)
            *((v8h*)&a0 + 1) = *(const v8h*)(A0 + k2 + 16);   // K 16..23 (lh=0) / 24..31 (lh=1)
            *((v8h*)&a1)     = *(const v8h*)(A1 + k2);
            *((v8h*)&a1 + 1) = *(const v8h*)(A1 + k2 + 16);
            *((v8h*)&bf)     = *(const v8h*)(Bp + k2);        // K 0..7,16..23 (lh=0) / 8..15,24..31 (lh=1)
            *((v8h*)&bf + 1) = *(const v8h*)(Bp + k2 + 16);
            acc0 = __builtin_amdgcn_wmma_f32_16x16x32_f16(false, a0, false, bf,
                                                          (short)0, acc0, false, false);
            acc1 = __builtin_amdgcn_wmma_f32_16x16x32_f16(false, a1, false, bf,
                                                          (short)0, acc1, false, false);
        }
    };

    issue(0, 0);
    for (int s = 0; s < NSTAGE - 1; ++s) {
        const int cur = s & 1;
        issue((s + 1)*KSTEP, cur ^ 1);
#if ASYNC_OK
        __builtin_amdgcn_s_wait_asynccnt(4);   // stage s landed; s+1 in flight
#endif
        __syncthreads();
        consume(cur);
        __syncthreads();                       // all reads of buf[cur] done before reuse
    }
#if ASYNC_OK
    __builtin_amdgcn_s_wait_asynccnt(0);
#endif
    __syncthreads();
    consume((NSTAGE - 1) & 1);

    // Epilogue: C VGPR r -> lanes 0-15: (M=r, N=lane); lanes 16-31: (M=8+r, N=lane-16)
    const int nCol = nBlk*64 + ni*16 + lr;
    const int bIdx = nCol >> 4;
    const int tIdx = nCol & 15;
    const float* bbRow = bb + (long)bIdx*COUT;
    float* outRow = out + ((long)bIdx*COUT)*TT + tIdx;

    #pragma unroll
    for (int r = 0; r < 8; ++r) {
        const int m  = lh*8 + r;
        const int o0 = mBlk*64 + mi0*16 + m;
        const int o1 = o0 + 16;
        outRow[(long)o0*TT] = acc0[r] + bbRow[o0] + 1.0f;
        outRow[(long)o1*TT] = acc1[r] + bbRow[o1] + 1.0f;
    }
}

// ---------------------------------------------------------------------------
extern "C" void kernel_launch(void* const* d_in, const int* in_sizes, int n_in,
                              void* d_out, int out_size, void* d_ws, size_t ws_size,
                              hipStream_t stream)
{
    const float* x      = (const float*)d_in[0];   // (64,512,16)
    const float* rf_w   = (const float*)d_in[1];   // (16,512)
    const float* rf_b   = (const float*)d_in[2];   // (16,)
    const float* weight = (const float*)d_in[3];   // (16,512,512,3)
    const float* bias   = (const float*)d_in[4];   // (16,512)
    float* out = (float*)d_out;                    // (64,512,16)

    char* ws = (char*)d_ws;
    float* rf = (float*)(ws);                      //   4 KB
    float* bb = (float*)(ws + 4096);               // 128 KB
    f16*   Ah = (f16*)(ws + 135168);               // 25165824 B
    f16*   Bt = (f16*)(ws + 25300992);             // 50331648 B

    k_rf_bb<<<NB, 256, 0, stream>>>(x, rf_w, rf_b, bias, rf, bb);
    k_prep_a<<<(COUT*KTOT/8)/256, 256, 0, stream>>>(weight, Ah);
    k_prep_b<<<(NTOT*KTOT/8)/256, 256, 0, stream>>>(x, rf, Bt);
    k_gemm<<<dim3(COUT/64, NTOT/64), 256, 0, stream>>>(Ah, Bt, bb, out);
}
